// DGTI_Model_35150012350942
// MI455X (gfx1250) — matrix operand order repaired and checked by reading, not compiled, not run on hardware
//
#include <hip/hip_runtime.h>
#include <hip/hip_bf16.h>
#include <math.h>

// Problem constants
constexpr int BB  = 2;    // batch
constexpr int TT  = 8;    // timesteps
constexpr int NN  = 512;  // nodes
constexpr int HH  = 64;   // hidden
constexpr int H1  = 4;    // heads, layer 1
constexpr int CC1 = 16;   // channels per head, layer 1
constexpr int NC  = 10;   // classes
constexpr float NEGV = -1e9f;

typedef __attribute__((ext_vector_type(16))) _Float16 v16h;
typedef __attribute__((ext_vector_type(8)))  float    v8f;

__device__ __forceinline__ float lrelu02(float x) { return x > 0.f ? x : 0.2f * x; }
__device__ __forceinline__ float eluf(float x)    { return x > 0.f ? x : (expf(x) - 1.f); }

// ---------------------------------------------------------------------------
// dyn_pos = relu(dyn_W + dyn_W^T)
__global__ void k_dynpos(const float* __restrict__ dynW, float* __restrict__ dyn_pos) {
    int idx = blockIdx.x * blockDim.x + threadIdx.x;
    int i = idx / NN, j = idx % NN;
    float v = dynW[i * NN + j] + dynW[j * NN + i];
    dyn_pos[idx] = v > 0.f ? v : 0.f;
}

// ---------------------------------------------------------------------------
// Layer-1 linear transforms: xl1/xr1 laid out (b, head, node, c)
__global__ void k_xlr1(const float* __restrict__ x_seq,
                       const float* __restrict__ Wl, const float* __restrict__ Wr,
                       float* __restrict__ xl1, float* __restrict__ xr1, int t) {
    int idx = blockIdx.x * blockDim.x + threadIdx.x;  // (b,h,n,c)
    int c = idx % CC1;
    int n = (idx / CC1) % NN;
    int h = (idx / (CC1 * NN)) % H1;
    int b = idx / (CC1 * NN * H1);
    float x = x_seq[(b * TT + t) * NN + n];
    int w = h * CC1 + c;
    xl1[idx] = x * Wl[w];
    xr1[idx] = x * Wr[w];
}

// ---------------------------------------------------------------------------
// GATv2 layer-1 scores with time-gated fused-adjacency mask
__global__ void k_e1(const float* __restrict__ xl1, const float* __restrict__ xr1,
                     const float* __restrict__ att,
                     const float* __restrict__ stat, const float* __restrict__ dyn_pos,
                     const float* __restrict__ reg_lambda,
                     float* __restrict__ e1, int t) {
    int idx = blockIdx.x * blockDim.x + threadIdx.x;  // (bh, i, j)
    int j  = idx % NN;
    int i  = (idx / NN) % NN;
    int bh = idx / (NN * NN);
    int h  = bh % H1;
    const float* xr = xr1 + (bh * NN + i) * CC1;
    const float* xl = xl1 + (bh * NN + j) * CC1;
    const float* at = att + h * CC1;
    float acc = 0.f;
#pragma unroll
    for (int c = 0; c < CC1; ++c) acc += at[c] * lrelu02(xr[c] + xl[c]);
    float lam = fmaxf(reg_lambda[0], 0.01f);
    float gt  = expf(-lam * (float)t);
    float fused = gt * stat[i * NN + j] + (1.f - gt) * dyn_pos[i * NN + j];
    bool m = (fused > 0.f) || (i == j);
    e1[idx] = m ? acc : NEGV;
}

// ---------------------------------------------------------------------------
// Row softmax over 512 entries, one wave32 per row, in place
__global__ void k_softmax512(float* __restrict__ data) {
    int wid  = blockIdx.x * (blockDim.x >> 5) + (threadIdx.x >> 5);
    int lane = threadIdx.x & 31;
    float* row = data + (size_t)wid * NN;
    float mx = -3.4e38f;
    for (int j = lane; j < NN; j += 32) mx = fmaxf(mx, row[j]);
#pragma unroll
    for (int o = 16; o > 0; o >>= 1) mx = fmaxf(mx, __shfl_xor(mx, o, 32));
    float sum = 0.f;
    for (int j = lane; j < NN; j += 32) { float e = expf(row[j] - mx); row[j] = e; sum += e; }
#pragma unroll
    for (int o = 16; o > 0; o >>= 1) sum += __shfl_xor(sum, o, 32);
    float inv = 1.f / sum;
    for (int j = lane; j < NN; j += 32) row[j] *= inv;
}

// ---------------------------------------------------------------------------
// Batched f32 GEMM via v_wmma_f32_16x16x32_f16: one wave32 per 16x16 C tile.
// A (MxK, lda), B (KxN, ldb), C (MxN, ldc); f32 in memory, f16 on WMMA A/B,
// f32 accumulate. Fragment layouts per CDNA5 ISA 7.12.2.
//
// A loads are vectorized: per K-step each lane issues 4x b128 for A (two
// contiguous 8-float runs: lane<16 -> K {0..7,16..23}, lane>=16 ->
// K {8..15,24..31}) and 16 scalar loads for B (column N=n0+(lane&15), K strided by ldb).
// All call sites guarantee 16B alignment (lda/ldb multiples of 16, k0%32==0,
// n0%16==0).
#define CVT4(dst, off, f4)                          \
    dst[(off) + 0] = (_Float16)(f4).x;              \
    dst[(off) + 1] = (_Float16)(f4).y;              \
    dst[(off) + 2] = (_Float16)(f4).z;              \
    dst[(off) + 3] = (_Float16)(f4).w;

__global__ void __launch_bounds__(32)
k_gemm(const float* __restrict__ A, const float* __restrict__ B, float* __restrict__ C,
       int K, int lda, int ldb, int ldc,
       long long sA, long long sB, long long sC) {
    const float* Ab = A + (long long)blockIdx.z * sA;
    const float* Bb = B + (long long)blockIdx.z * sB;
    float*       Cb = C + (long long)blockIdx.z * sC;
    int m0 = blockIdx.x * 16, n0 = blockIdx.y * 16;
    int lane = threadIdx.x & 31;
    bool hi = lane >= 16;
    int arow = m0 + (lane & 15);
    const float* Arow = Ab + (size_t)arow * lda;
    int asel = hi ? 8 : 0;
    v8f acc = {};
    for (int k0 = 0; k0 < K; k0 += 32) {
        // A fragment: elements 0..7 <- K=k0+asel..+7 ; 8..15 <- K=k0+16+asel..+7
        const float4* pa0 = (const float4*)(Arow + k0 + asel);
        const float4* pa1 = (const float4*)(Arow + k0 + 16 + asel);
        float4 a0 = pa0[0], a1 = pa0[1];
        float4 a2 = pa1[0], a3 = pa1[1];
        // B fragment: column N = n0+(lane&15); elements 0..7 <- K=k0+asel..+7 ; 8..15 <- K=k0+16+asel..+7
        const float* pb = Bb + (size_t)(k0 + asel) * ldb + n0 + (lane & 15);
        v16h a, b;
        CVT4(a, 0, a0)  CVT4(a, 4, a1)  CVT4(a, 8, a2)  CVT4(a, 12, a3)
        for (int s = 0; s < 8; ++s) {
            b[s]     = (_Float16)pb[(size_t)s * ldb];
            b[s + 8] = (_Float16)pb[(size_t)(16 + s) * ldb];
        }
        acc = __builtin_amdgcn_wmma_f32_16x16x32_f16(false, a, false, b,
                                                     (short)0, acc, false, false);
    }
    int crow = m0 + (hi ? 8 : 0);
    int ccol = n0 + (lane & 15);
#pragma unroll
    for (int v = 0; v < 8; ++v)
        Cb[(crow + v) * ldc + ccol] = acc[v];
}

// ---------------------------------------------------------------------------
// h1[b,i,h*16+c] = elu(out1[b,h,i,c] + bias[h*16+c])
__global__ void k_h1(const float* __restrict__ out1, const float* __restrict__ bias,
                     float* __restrict__ h1) {
    int idx = blockIdx.x * blockDim.x + threadIdx.x;  // (b,h,i,c)
    int c = idx % CC1;
    int i = (idx / CC1) % NN;
    int h = (idx / (CC1 * NN)) % H1;
    int b = idx / (CC1 * NN * H1);
    float v = out1[idx] + bias[h * CC1 + c];
    h1[(b * NN + i) * HH + h * CC1 + c] = eluf(v);
}

// ---------------------------------------------------------------------------
// GATv2 layer-2 scores (single head, C=64)
__global__ void k_e2(const float* __restrict__ xl2, const float* __restrict__ xr2,
                     const float* __restrict__ att2,
                     const float* __restrict__ stat, const float* __restrict__ dyn_pos,
                     const float* __restrict__ reg_lambda,
                     float* __restrict__ e2, int t) {
    int idx = blockIdx.x * blockDim.x + threadIdx.x;  // (b, i, j)
    int j = idx % NN;
    int i = (idx / NN) % NN;
    int b = idx / (NN * NN);
    const float* xr = xr2 + (b * NN + i) * HH;
    const float* xl = xl2 + (b * NN + j) * HH;
    float acc = 0.f;
#pragma unroll 8
    for (int c = 0; c < HH; ++c) acc += att2[c] * lrelu02(xr[c] + xl[c]);
    float lam = fmaxf(reg_lambda[0], 0.01f);
    float gt  = expf(-lam * (float)t);
    float fused = gt * stat[i * NN + j] + (1.f - gt) * dyn_pos[i * NN + j];
    bool m = (fused > 0.f) || (i == j);
    e2[idx] = m ? acc : NEGV;
}

// h2 = elu(h2 + bias), in place
__global__ void k_h2(float* __restrict__ h2, const float* __restrict__ bias) {
    int idx = blockIdx.x * blockDim.x + threadIdx.x;  // (b,i,c)
    int c = idx % HH;
    h2[idx] = eluf(h2[idx] + bias[c]);
}

// graph readout: reps[b,t,:] = mean_i h2[b,i,:]
__global__ void k_mean(const float* __restrict__ h2, float* __restrict__ reps, int t) {
    int tid = threadIdx.x;             // 128 threads
    int b = tid / HH, u = tid % HH;
    float s = 0.f;
    for (int i = 0; i < NN; ++i) s += h2[(b * NN + i) * HH + u];
    reps[(b * TT + t) * HH + u] = s * (1.f / (float)NN);
}

// ---------------------------------------------------------------------------
// One GRU layer over (BB,TT,HH); xin may be global or shared (generic ptr).
__device__ __forceinline__ void gru_layer(const float* xin,
                                          const float* __restrict__ Wih,
                                          const float* __restrict__ Whh,
                                          const float* __restrict__ bih,
                                          const float* __restrict__ bhh,
                                          float* hsh, float* yout,
                                          int b, int u, int tid) {
    hsh[tid] = 0.f;
    __syncthreads();
    for (int t = 0; t < TT; ++t) {
        const float* xt = xin + (b * TT + t) * HH;
        float ir = bih[u], iz = bih[HH + u], in = bih[2 * HH + u];
        float hr = bhh[u], hz = bhh[HH + u], hn = bhh[2 * HH + u];
        for (int k = 0; k < HH; ++k) {
            float xv = xt[k], hv = hsh[b * HH + k];
            const float* wi = Wih + k * 3 * HH;
            const float* wh = Whh + k * 3 * HH;
            ir += xv * wi[u];        iz += xv * wi[HH + u];   in += xv * wi[2 * HH + u];
            hr += hv * wh[u];        hz += hv * wh[HH + u];   hn += hv * wh[2 * HH + u];
        }
        float r = 1.f / (1.f + expf(-(ir + hr)));
        float z = 1.f / (1.f + expf(-(iz + hz)));
        float n = tanhf(in + r * hn);
        float hnew = (1.f - z) * n + z * hsh[b * HH + u];
        __syncthreads();
        hsh[b * HH + u] = hnew;
        yout[(b * TT + t) * HH + u] = hnew;
        __syncthreads();
    }
}

__global__ void __launch_bounds__(128)
k_gru(const float* __restrict__ reps,
      const float* __restrict__ Wih0, const float* __restrict__ Whh0,
      const float* __restrict__ bih0, const float* __restrict__ bhh0,
      const float* __restrict__ Wih1, const float* __restrict__ Whh1,
      const float* __restrict__ bih1, const float* __restrict__ bhh1,
      float* __restrict__ gru_out) {
    __shared__ float hsh[BB * HH];
    __shared__ float ys0[BB * TT * HH];
    int tid = threadIdx.x;
    int b = tid / HH, u = tid % HH;
    gru_layer(reps, Wih0, Whh0, bih0, bhh0, hsh, ys0, b, u, tid);
    gru_layer(ys0,  Wih1, Whh1, bih1, bhh1, hsh, gru_out, b, u, tid);
}

// ---------------------------------------------------------------------------
// Temporal attention + classifier head; writes all 44 output floats.
__global__ void __launch_bounds__(128)
k_head(const float* __restrict__ gru_out,
       const float* __restrict__ attn_W, const float* __restrict__ attn_b,
       const float* __restrict__ W1, const float* __restrict__ b1,
       const float* __restrict__ lg, const float* __restrict__ lb,
       const float* __restrict__ W2, const float* __restrict__ b2,
       const float* __restrict__ reg_lambda,
       float* __restrict__ out) {
    __shared__ float sc[BB * TT];
    __shared__ float fin[BB * HH];
    __shared__ float zsh[BB * HH];
    __shared__ float stat2[BB * 2];
    int tid = threadIdx.x;            // 128
    int b = tid / HH, u = tid % HH;

    if (tid < BB * TT) {
        int bb = tid / TT, t = tid % TT;
        float s = attn_b[0];
        for (int k = 0; k < HH; ++k) s += gru_out[(bb * TT + t) * HH + k] * attn_W[k];
        sc[tid] = s;
    }
    __syncthreads();
    if (tid < BB) {
        float mx = -3.4e38f;
        for (int t = 0; t < TT; ++t) mx = fmaxf(mx, sc[tid * TT + t]);
        float sm = 0.f;
        for (int t = 0; t < TT; ++t) { float e = expf(sc[tid * TT + t] - mx); sc[tid * TT + t] = e; sm += e; }
        float inv = 1.f / sm;
        for (int t = 0; t < TT; ++t) sc[tid * TT + t] *= inv;
    }
    __syncthreads();
    float f = 0.f;
    for (int t = 0; t < TT; ++t) f += sc[b * TT + t] * gru_out[(b * TT + t) * HH + u];
    fin[tid] = f;
    __syncthreads();
    float z = b1[u];
    for (int k = 0; k < HH; ++k) z += fin[b * HH + k] * W1[k * HH + u];
    zsh[tid] = z;
    __syncthreads();
    if (u == 0) {
        float mu = 0.f;
        for (int k = 0; k < HH; ++k) mu += zsh[b * HH + k];
        mu *= (1.f / HH);
        float var = 0.f;
        for (int k = 0; k < HH; ++k) { float d = zsh[b * HH + k] - mu; var += d * d; }
        var *= (1.f / HH);
        stat2[b * 2] = mu; stat2[b * 2 + 1] = var;
    }
    __syncthreads();
    float zn = (z - stat2[b * 2]) * rsqrtf(stat2[b * 2 + 1] + 1e-5f) * lg[u] + lb[u];
    float g = 0.5f * zn * (1.f + erff(zn * 0.70710678118f));  // exact GELU
    zsh[tid] = g;
    __syncthreads();
    if (tid < BB * NC) {
        int bb = tid / NC, c = tid % NC;
        float acc = b2[c];
        for (int k = 0; k < HH; ++k) acc += zsh[bb * HH + k] * W2[k * NC + c];
        out[bb * NC + c] = acc;                       // logits
    }
    float lam = fmaxf(reg_lambda[0], 0.01f);
    if (tid < TT)       out[BB * NC + tid] = expf(-lam * (float)tid);   // gts
    if (tid < BB * TT)  out[BB * NC + TT + tid] = sc[tid];              // att_scores
}

// ---------------------------------------------------------------------------
extern "C" void kernel_launch(void* const* d_in, const int* in_sizes, int n_in,
                              void* d_out, int out_size, void* d_ws, size_t ws_size,
                              hipStream_t stream) {
    (void)in_sizes; (void)n_in; (void)out_size; (void)ws_size;
    const float* x_seq   = (const float*)d_in[0];
    const float* stat    = (const float*)d_in[1];
    const float* dynW    = (const float*)d_in[2];
    const float* lam     = (const float*)d_in[3];
    const float* g1_Wl   = (const float*)d_in[4];
    const float* g1_Wr   = (const float*)d_in[5];
    const float* g1_att  = (const float*)d_in[6];
    const float* g1_b    = (const float*)d_in[7];
    const float* g2_Wl   = (const float*)d_in[8];
    const float* g2_Wr   = (const float*)d_in[9];
    const float* g2_att  = (const float*)d_in[10];
    const float* g2_b    = (const float*)d_in[11];
    const float* Wih0    = (const float*)d_in[12];
    const float* Whh0    = (const float*)d_in[13];
    const float* bih0    = (const float*)d_in[14];
    const float* bhh0    = (const float*)d_in[15];
    const float* Wih1    = (const float*)d_in[16];
    const float* Whh1    = (const float*)d_in[17];
    const float* bih1    = (const float*)d_in[18];
    const float* bhh1    = (const float*)d_in[19];
    const float* attn_W  = (const float*)d_in[20];
    const float* attn_b  = (const float*)d_in[21];
    const float* cls_W1  = (const float*)d_in[22];
    const float* cls_b1  = (const float*)d_in[23];
    const float* ln_g    = (const float*)d_in[24];
    const float* ln_b    = (const float*)d_in[25];
    const float* cls_W2  = (const float*)d_in[26];
    const float* cls_b2  = (const float*)d_in[27];
    float* outp = (float*)d_out;

    // Workspace carve-up (floats)
    float* ws = (float*)d_ws;
    float* dyn_pos = ws;                 ws += NN * NN;            // 262144
    float* xl1     = ws;                 ws += BB * H1 * NN * CC1; // 65536
    float* xr1     = ws;                 ws += BB * H1 * NN * CC1;
    float* e1      = ws;                 ws += BB * H1 * NN * NN;  // 2097152
    float* out1    = ws;                 ws += BB * H1 * NN * CC1;
    float* h1      = ws;                 ws += BB * NN * HH;
    float* xl2     = ws;                 ws += BB * NN * HH;
    float* xr2     = ws;                 ws += BB * NN * HH;
    float* e2      = ws;                 ws += BB * NN * NN;       // 524288
    float* h2      = ws;                 ws += BB * NN * HH;
    float* reps    = ws;                 ws += BB * TT * HH;
    float* gru_out = ws;                 ws += BB * TT * HH;

    k_dynpos<<<(NN * NN) / 256, 256, 0, stream>>>(dynW, dyn_pos);

    for (int t = 0; t < TT; ++t) {
        k_xlr1<<<(BB * H1 * NN * CC1) / 256, 256, 0, stream>>>(x_seq, g1_Wl, g1_Wr, xl1, xr1, t);
        k_e1<<<(BB * H1 * NN * NN) / 256, 256, 0, stream>>>(xl1, xr1, g1_att, stat, dyn_pos, lam, e1, t);
        k_softmax512<<<(BB * H1 * NN) / 8, 256, 0, stream>>>(e1);
        // out1[bh] (512x16) = alpha1[bh] (512x512) @ xl1[bh] (512x16)
        k_gemm<<<dim3(NN / 16, 1, BB * H1), 32, 0, stream>>>(
            e1, xl1, out1, NN, NN, CC1, CC1,
            (long long)NN * NN, (long long)NN * CC1, (long long)NN * CC1);
        k_h1<<<(BB * H1 * NN * CC1) / 256, 256, 0, stream>>>(out1, g1_b, h1);
        // xl2/xr2[b] (512x64) = h1[b] (512x64) @ W (64x64)
        k_gemm<<<dim3(NN / 16, HH / 16, BB), 32, 0, stream>>>(
            h1, g2_Wl, xl2, HH, HH, HH, HH,
            (long long)NN * HH, 0LL, (long long)NN * HH);
        k_gemm<<<dim3(NN / 16, HH / 16, BB), 32, 0, stream>>>(
            h1, g2_Wr, xr2, HH, HH, HH, HH,
            (long long)NN * HH, 0LL, (long long)NN * HH);
        k_e2<<<(BB * NN * NN) / 256, 256, 0, stream>>>(xl2, xr2, g2_att, stat, dyn_pos, lam, e2, t);
        k_softmax512<<<(BB * NN) / 8, 256, 0, stream>>>(e2);
        // h2[b] (512x64) = alpha2[b] (512x512) @ xl2[b] (512x64)
        k_gemm<<<dim3(NN / 16, HH / 16, BB), 32, 0, stream>>>(
            e2, xl2, h2, NN, NN, HH, HH,
            (long long)NN * NN, (long long)NN * HH, (long long)NN * HH);
        k_h2<<<(BB * NN * HH) / 256, 256, 0, stream>>>(h2, g2_b);
        k_mean<<<1, 128, 0, stream>>>(h2, reps, t);
    }

    k_gru<<<1, 128, 0, stream>>>(reps, Wih0, Whh0, bih0, bhh0,
                                 Wih1, Whh1, bih1, bhh1, gru_out);
    k_head<<<1, 128, 0, stream>>>(gru_out, attn_W, attn_b, cls_W1, cls_b1,
                                  ln_g, ln_b, cls_W2, cls_b2, lam, outp);
}
